// BatteryKAN_41875931136529
// MI455X (gfx1250) — compile-verified, matrix operand order checked by reading, not run on hardware
//
#include <hip/hip_runtime.h>

// ---------- types ----------
typedef unsigned int u32;
typedef unsigned long long u64;
typedef __attribute__((ext_vector_type(2)))  u32      u32x2;
typedef __attribute__((ext_vector_type(4)))  u32      u32x4;
typedef __attribute__((ext_vector_type(16))) _Float16 v16h;
typedef __attribute__((ext_vector_type(8)))  float    v8f;

// Dimensions of the KAN (reference: LAYERS = [512, 1024, 1024, 256], B = 8192)
#define BATCH   8192
#define EXPAND  12          // [silu, 11 spline bases] per input feature

// gfx1250 async Global->LDS copy path (Tensor/async data movement, ASYNCcnt).
// Flip to 0 to fall back to global_load_b128 + ds_store_b128 staging.
#define USE_ASYNC_LDS 1

__device__ __forceinline__ unsigned short f2h(float f) {
    _Float16 h = (_Float16)f;
    return __builtin_bit_cast(unsigned short, h);
}

__device__ __forceinline__ float bspline_core(float u) {
    float t2 = 2.0f - u;
    float a  = t2 * t2 * t2;
    float t1 = 1.0f - u;
    float b  = t1 * t1 * t1;
    float v1 = (a - 4.0f * b) * (1.0f / 6.0f);
    float v2 = a * (1.0f / 6.0f);
    return (u < 1.0f) ? v1 : ((u < 2.0f) ? v2 : 0.0f);
}

#if USE_ASYNC_LDS
// One per-lane 16B async copy: LDS[lds_off] = MEM[gaddr]; tracked by ASYNCcnt.
__device__ __forceinline__ void async_ld16(unsigned lds_off, const void* gaddr) {
    asm volatile("global_load_async_to_lds_b128 %0, %1, off"
                 :: "v"(lds_off), "v"((u64)(uintptr_t)gaddr)
                 : "memory");
}
template <int N>
__device__ __forceinline__ void wait_asynccnt() {
    asm volatile("s_wait_asynccnt %0" :: "i"(N) : "memory");
}
#endif

// ---------- expand: x[b,i] -> A[b, i*12 + {0..11}] = [silu(x), bases] (fp16) ----------
__global__ void kan_expand(const float* __restrict__ x,
                           unsigned short* __restrict__ A,
                           int total /* B*IN */) {
    int idx = blockIdx.x * blockDim.x + threadIdx.x;
    if (idx >= total) return;
    float v   = x[idx];
    float sig = 1.0f / (1.0f + __expf(-v));
    unsigned short h[12];
    h[0] = f2h(v * sig);
    #pragma unroll
    for (int k = 0; k < 11; ++k) {
        float c = -1.25f + 0.25f * (float)k;       // CENTERS = linspace(-1.25, 1.25, 11)
        float u = fabsf(v - c) * 4.0f;             // / GRID_STEP (0.25)
        h[1 + k] = f2h(bspline_core(u));
    }
    u32x2* dst = reinterpret_cast<u32x2*>(A + (size_t)idx * EXPAND); // 24B, 8B aligned
    u32x2 w0, w1, w2;
    w0.x = (u32)h[0]  | ((u32)h[1]  << 16);  w0.y = (u32)h[2]  | ((u32)h[3]  << 16);
    w1.x = (u32)h[4]  | ((u32)h[5]  << 16);  w1.y = (u32)h[6]  | ((u32)h[7]  << 16);
    w2.x = (u32)h[8]  | ((u32)h[9]  << 16);  w2.y = (u32)h[10] | ((u32)h[11] << 16);
    dst[0] = w0; dst[1] = w1; dst[2] = w2;
}

// ---------- pack: [base_w | spline_w] -> W[o, i*12 + {0..11}] (fp16) ----------
__global__ void kan_pack_w(const float* __restrict__ bw,
                           const float* __restrict__ sw,
                           unsigned short* __restrict__ W,
                           int total /* OUT*IN */) {
    int idx = blockIdx.x * blockDim.x + threadIdx.x;
    if (idx >= total) return;
    unsigned short h[12];
    h[0] = f2h(bw[idx]);
    const float* s = sw + (size_t)idx * 11;
    #pragma unroll
    for (int k = 0; k < 11; ++k) h[1 + k] = f2h(s[k]);
    u32x2* dst = reinterpret_cast<u32x2*>(W + (size_t)idx * EXPAND);
    u32x2 w0, w1, w2;
    w0.x = (u32)h[0]  | ((u32)h[1]  << 16);  w0.y = (u32)h[2]  | ((u32)h[3]  << 16);
    w1.x = (u32)h[4]  | ((u32)h[5]  << 16);  w1.y = (u32)h[6]  | ((u32)h[7]  << 16);
    w2.x = (u32)h[8]  | ((u32)h[9]  << 16);  w2.y = (u32)h[10] | ((u32)h[11] << 16);
    dst[0] = w0; dst[1] = w1; dst[2] = w2;
}

// ---------- fused GEMM + clip:  Y[b,o] = clip( A[b,:] . W[o,:], -1, 1 ) ----------
// A: [BATCH, K] fp16 row-major, W: [N, K] fp16 row-major, Y: [BATCH, N] fp32
// Block: 256 threads (8 waves), tile 128(M) x 128(N), K stepped by 32,
// double-buffered LDS fed by async Global->LDS DMA (ASYNCcnt).
// Wave (wid) covers rows wm..wm+63, cols wn..wn+31 -> 4x2 WMMA accumulators.
__global__ __launch_bounds__(256)
void kan_wmma_gemm(const unsigned short* __restrict__ A,
                   const unsigned short* __restrict__ W,
                   float* __restrict__ Y,
                   int N, int K) {
    __shared__ unsigned short lA[2][128 * 32];   // 2 x 8 KB
    __shared__ unsigned short lW[2][128 * 32];   // 2 x 8 KB

    const int tid  = threadIdx.x;
    const int wid  = tid >> 5;
    const int lane = tid & 31;
    const int l15  = lane & 15;
    const int sel  = lane >> 4;               // 0: K{0-7,16-23}, 1: K{8-15,24-31}
    const int m0   = blockIdx.x * 128;
    const int n0   = blockIdx.y * 128;
    const int wm   = (wid & 1) * 64;
    const int wn   = (wid >> 1) * 32;

    v8f acc[4][2];
    #pragma unroll
    for (int mt = 0; mt < 4; ++mt)
        #pragma unroll
        for (int nt = 0; nt < 2; ++nt) {
            v8f z = {0.f, 0.f, 0.f, 0.f, 0.f, 0.f, 0.f, 0.f};
            acc[mt][nt] = z;
        }

    // Tile staging: 512 x 16B chunks per 128x32 fp16 tile; thread -> chunks {tid, tid+256}.
    const int c0 = tid, c1 = tid + 256;
    const int r0 = c0 >> 2, q0 = c0 & 3;
    const int r1 = c1 >> 2, q1 = c1 & 3;
    const unsigned short* gA0 = A + (size_t)(m0 + r0) * K + q0 * 8;
    const unsigned short* gA1 = A + (size_t)(m0 + r1) * K + q1 * 8;
    const unsigned short* gW0 = W + (size_t)(n0 + r0) * K + q0 * 8;
    const unsigned short* gW1 = W + (size_t)(n0 + r1) * K + q1 * 8;
#if USE_ASYNC_LDS
    const unsigned ldsA = (unsigned)(uintptr_t)(&lA[0][0]);
    const unsigned ldsW = (unsigned)(uintptr_t)(&lW[0][0]);
#endif

    union Frag { u32x4 q[2]; v16h v; };

    // Stage tile at K-offset kt into buffer buf (4 async instrs per wave).
    auto stage = [&](int buf, int kt) {
#if USE_ASYNC_LDS
        async_ld16(ldsA + (unsigned)(buf * 8192 + c0 * 16), gA0 + kt);
        async_ld16(ldsA + (unsigned)(buf * 8192 + c1 * 16), gA1 + kt);
        async_ld16(ldsW + (unsigned)(buf * 8192 + c0 * 16), gW0 + kt);
        async_ld16(ldsW + (unsigned)(buf * 8192 + c1 * 16), gW1 + kt);
#else
        reinterpret_cast<u32x4*>(&lA[buf][0])[c0] = *reinterpret_cast<const u32x4*>(gA0 + kt);
        reinterpret_cast<u32x4*>(&lA[buf][0])[c1] = *reinterpret_cast<const u32x4*>(gA1 + kt);
        reinterpret_cast<u32x4*>(&lW[buf][0])[c0] = *reinterpret_cast<const u32x4*>(gW0 + kt);
        reinterpret_cast<u32x4*>(&lW[buf][0])[c1] = *reinterpret_cast<const u32x4*>(gW1 + kt);
#endif
    };

    const int nsteps = K >> 5;           // K / 32
    stage(0, 0);

    for (int s = 0; s < nsteps; ++s) {
        const int cur = s & 1;
        if (s + 1 < nsteps) {
            stage(cur ^ 1, (s + 1) << 5);            // overlap next fetch with this math
            // Far prefetch (2 tiles ahead) to prime GL2.
            if (s + 2 < nsteps) {
                __builtin_prefetch(gA0 + ((s + 2) << 5), 0, 1);
                __builtin_prefetch(gW0 + ((s + 2) << 5), 0, 1);
            }
#if USE_ASYNC_LDS
            wait_asynccnt<4>();          // in-order: current tile's 4 copies done
#endif
        } else {
#if USE_ASYNC_LDS
            wait_asynccnt<0>();
#endif
        }
        __syncthreads();                 // all waves' copies of buffer `cur` visible

        // Assemble fragments per ISA 16-bit operand layout:
        // lanes 0-15 hold row l15 with K{0-7,16-23}; lanes 16-31 K{8-15,24-31}.
        Frag af[4], bfr[2];
        #pragma unroll
        for (int mt = 0; mt < 4; ++mt) {
            const u32x4* p = reinterpret_cast<const u32x4*>(&lA[cur][(wm + mt * 16 + l15) * 32]);
            af[mt].q[0] = p[sel];
            af[mt].q[1] = p[2 + sel];
        }
        #pragma unroll
        for (int nt = 0; nt < 2; ++nt) {
            const u32x4* p = reinterpret_cast<const u32x4*>(&lW[cur][(wn + nt * 16 + l15) * 32]);
            bfr[nt].q[0] = p[sel];
            bfr[nt].q[1] = p[2 + sel];
        }

        #pragma unroll
        for (int mt = 0; mt < 4; ++mt)
            #pragma unroll
            for (int nt = 0; nt < 2; ++nt)
                acc[mt][nt] = __builtin_amdgcn_wmma_f32_16x16x32_f16(
                    /*neg_a=*/false, af[mt].v,
                    /*neg_b=*/false, bfr[nt].v,
                    /*c_mod=*/(short)0, acc[mt][nt],
                    /*reuse_a=*/false, /*reuse_b=*/false);

        __syncthreads();                 // readers done before buffer `cur` is re-filled
    }

    // Epilogue: clip to [-1, 1] and store.
    // C layout: lanes 0-15 -> M = r, N = lane; lanes 16-31 -> M = 8 + r, N = lane - 16.
    #pragma unroll
    for (int mt = 0; mt < 4; ++mt) {
        #pragma unroll
        for (int nt = 0; nt < 2; ++nt) {
            const int col = n0 + wn + nt * 16 + l15;
            #pragma unroll
            for (int r = 0; r < 8; ++r) {
                const int row = m0 + wm + mt * 16 + sel * 8 + r;
                float v = acc[mt][nt][r];
                v = fminf(1.0f, fmaxf(-1.0f, v));
                Y[(size_t)row * N + col] = v;
            }
        }
    }
}

// ---------- launcher ----------
extern "C" void kernel_launch(void* const* d_in, const int* in_sizes, int n_in,
                              void* d_out, int out_size, void* d_ws, size_t ws_size,
                              hipStream_t stream) {
    (void)in_sizes; (void)n_in; (void)out_size; (void)ws_size;
    const float* x   = (const float*)d_in[0];
    const float* bw0 = (const float*)d_in[1];
    const float* sw0 = (const float*)d_in[2];
    const float* bw1 = (const float*)d_in[3];
    const float* sw1 = (const float*)d_in[4];
    const float* bw2 = (const float*)d_in[5];
    const float* sw2 = (const float*)d_in[6];
    float* out = (float*)d_out;

    // K dims: 512*12 = 6144, 1024*12 = 12288
    const int K0 = 6144, K1 = 12288, K2 = 12288;

    char* ws = (char*)d_ws;
    const size_t A_OFF  = 0;                                        // 8192*12288*2 = 201326592
    const size_t X1_OFF = A_OFF  + (size_t)BATCH * 12288 * 2;       // 8192*1024*4  = 33554432
    const size_t X2_OFF = X1_OFF + (size_t)BATCH * 1024 * 4;
    const size_t W0_OFF = X2_OFF + (size_t)BATCH * 1024 * 4;        // 1024*6144*2
    const size_t W1_OFF = W0_OFF + (size_t)1024 * K0 * 2;           // 1024*12288*2
    const size_t W2_OFF = W1_OFF + (size_t)1024 * K1 * 2;           // 256*12288*2

    unsigned short* Abuf = (unsigned short*)(ws + A_OFF);
    float*          x1   = (float*)(ws + X1_OFF);
    float*          x2   = (float*)(ws + X2_OFF);
    unsigned short* W0   = (unsigned short*)(ws + W0_OFF);
    unsigned short* W1   = (unsigned short*)(ws + W1_OFF);
    unsigned short* W2   = (unsigned short*)(ws + W2_OFF);

    // Pack folded weights (fp32 -> fp16, [out, 12*in])
    kan_pack_w<<<dim3((1024 * 512  + 255) / 256), 256, 0, stream>>>(bw0, sw0, W0, 1024 * 512);
    kan_pack_w<<<dim3((1024 * 1024 + 255) / 256), 256, 0, stream>>>(bw1, sw1, W1, 1024 * 1024);
    kan_pack_w<<<dim3((256  * 1024 + 255) / 256), 256, 0, stream>>>(bw2, sw2, W2, 256 * 1024);

    // Layer 0: 512 -> 1024
    kan_expand<<<dim3((BATCH * 512) / 256), 256, 0, stream>>>(x, Abuf, BATCH * 512);
    kan_wmma_gemm<<<dim3(BATCH / 128, 1024 / 128), 256, 0, stream>>>(Abuf, W0, x1, 1024, K0);

    // Layer 1: 1024 -> 1024
    kan_expand<<<dim3((BATCH * 1024) / 256), 256, 0, stream>>>(x1, Abuf, BATCH * 1024);
    kan_wmma_gemm<<<dim3(BATCH / 128, 1024 / 128), 256, 0, stream>>>(Abuf, W1, x2, 1024, K1);

    // Layer 2: 1024 -> 256
    kan_expand<<<dim3((BATCH * 1024) / 256), 256, 0, stream>>>(x2, Abuf, BATCH * 1024);
    kan_wmma_gemm<<<dim3(BATCH / 128, 256 / 128), 256, 0, stream>>>(Abuf, W2, out, 256, K2);
}
